// LSTMCellManual_14654428413974
// MI455X (gfx1250) — matrix operand order repaired and checked by reading, not compiled, not run on hardware
//
#include <hip/hip_runtime.h>

// LSTM cell, B=4096, D_IN=D_H=1024, fused 8-GEMM (4 gates over K=2048) +
// activation epilogue. bf16 WMMA (v_wmma_f32_16x16x32_bf16), fp32 accumulate.

#define BATCH   4096
#define DH      1024
#define KTOT    2048
#define KSTEP   32
#define KITERS  (KTOT / KSTEP)

#define TILE_M  128
#define TILE_N  64
#define LDSTR   40   // halves per row/col in LDS: 80B, 16B-aligned, conflict-free

#define ROWSTRIDE_B (32u * DH * 4u)   // 131072 B: 32 rows of fp32; fits 24-bit ioffset

typedef __bf16 bf16_t;
typedef bf16_t bf16x16 __attribute__((ext_vector_type(16)));
typedef float  v8f     __attribute__((ext_vector_type(8)));

union Frag { uint4 u[2]; bf16x16 v; };

// Two fp32 -> packed bf16x2: one v_add each (round-to-nearest) and a single
// v_perm_b32 byte-select (high halves of a then b).
__device__ __forceinline__ unsigned pack_bf16(float f0, float f1) {
    unsigned a = __builtin_bit_cast(unsigned, f0) + 0x8000u;
    unsigned b = __builtin_bit_cast(unsigned, f1) + 0x8000u;
    return __builtin_amdgcn_perm(b, a, 0x07060302u);
}

__device__ __forceinline__ float fsigmoid(float x) { return 1.0f / (1.0f + __expf(-x)); }
__device__ __forceinline__ float ftanh(float x)    { return 2.0f / (1.0f + __expf(-2.0f * x)) - 1.0f; }

__global__ __launch_bounds__(256)
void lstm_cell_wmma(
    const float* __restrict__ x, const float* __restrict__ h, const float* __restrict__ c,
    const float* __restrict__ Wi0, const float* __restrict__ Wi1,
    const float* __restrict__ Wi2, const float* __restrict__ Wi3,
    const float* __restrict__ Wh0, const float* __restrict__ Wh1,
    const float* __restrict__ Wh2, const float* __restrict__ Wh3,
    const float* __restrict__ bi0, const float* __restrict__ bi1,
    const float* __restrict__ bi2, const float* __restrict__ bi3,
    const float* __restrict__ bh0, const float* __restrict__ bh1,
    const float* __restrict__ bh2, const float* __restrict__ bh3,
    float* __restrict__ out)
{
    __shared__ unsigned short lsA[2][TILE_M * LDSTR];          // 20 KB
    __shared__ unsigned short lsB[2][4][TILE_N * LDSTR];       // 40 KB

    const float* Wi[4] = {Wi0, Wi1, Wi2, Wi3};
    const float* Wh[4] = {Wh0, Wh1, Wh2, Wh3};
    const float* bi[4] = {bi0, bi1, bi2, bi3};
    const float* bh[4] = {bh0, bh1, bh2, bh3};

    const int tid  = threadIdx.x;
    const int lane = tid & 31;
    const int wave = tid >> 5;
    const int l    = lane & 15;
    const int hi   = lane >> 4;
    const int wm   = wave & 3;      // 4 waves along M
    const int wn   = wave >> 2;     // 2 waves along N
    const int m0   = blockIdx.y * TILE_M;
    const int n0   = blockIdx.x * TILE_N;

    // ---- per-thread invariant offsets -----------------------------------
    // Staging decomposition: thread handles chunk (rowT + 32*j, c4T).
    const unsigned rowT = (unsigned)tid >> 3;         // 0..31
    const unsigned c4T  = ((unsigned)tid & 7) * 4;    // float index 0..28
    const unsigned gOffA0 = (((unsigned)m0 + rowT) * DH + c4T) * 4u;  // bytes
    const unsigned gOffB0 = (((unsigned)n0 + rowT) * DH + c4T) * 4u;  // bytes
    const int ldsOffA0 = (int)(rowT * LDSTR + c4T);   // halves; +j*32*LDSTR
    const int ldsOffB0 = ldsOffA0;

    // Fragment gather offsets (halves).
    int fragA[2], fragB[2];
    #pragma unroll
    for (int mA = 0; mA < 2; ++mA)
        fragA[mA] = (wm * 32 + mA * 16 + l) * LDSTR + 16 * hi;
    #pragma unroll
    for (int nB = 0; nB < 2; ++nB)
        fragB[nB] = (wn * 32 + nB * 16 + l) * LDSTR + 16 * hi;

    v8f acc[4][2][2];
    #pragma unroll
    for (int g = 0; g < 4; ++g)
        #pragma unroll
        for (int a = 0; a < 2; ++a)
            #pragma unroll
            for (int b = 0; b < 2; ++b)
                acc[g][a][b] = (v8f)0.0f;

    float bsum[4][2];
    #pragma unroll
    for (int g = 0; g < 4; ++g)
        #pragma unroll
        for (int nB = 0; nB < 2; ++nB) {
            int ng = n0 + wn * 32 + nB * 16 + l;
            bsum[g][nB] = bi[g][ng] + bh[g][ng];
        }

    // ---- software-pipelined K loop --------------------------------------
    float4 astg[4];
    float4 bstg[4][2];

    auto load_stage = [&](int kt) {
        const int kb = kt * KSTEP;
        const int ko = (kb < DH) ? kb : (kb - DH);
        // One vector address per group; row deltas go in the 24-bit ioffset.
        const char* pa = (const char*)(((kb < DH) ? x : h) + ko) + gOffA0;
        #pragma unroll
        for (int j = 0; j < 4; ++j)
            astg[j] = *(const float4*)(pa + (size_t)(j * ROWSTRIDE_B));
        #pragma unroll
        for (int g = 0; g < 4; ++g) {
            const char* pb = (const char*)(((kb < DH) ? Wi[g] : Wh[g]) + ko) + gOffB0;
            #pragma unroll
            for (int j = 0; j < 2; ++j)
                bstg[g][j] = *(const float4*)(pb + (size_t)(j * ROWSTRIDE_B));
        }
    };

    auto store_stage = [&](int buf) {
        #pragma unroll
        for (int j = 0; j < 4; ++j) {
            uint2 pk;
            pk.x = pack_bf16(astg[j].x, astg[j].y);
            pk.y = pack_bf16(astg[j].z, astg[j].w);
            *(uint2*)&lsA[buf][ldsOffA0 + j * 32 * LDSTR] = pk;
        }
        #pragma unroll
        for (int g = 0; g < 4; ++g)
            #pragma unroll
            for (int j = 0; j < 2; ++j) {
                uint2 pk;
                pk.x = pack_bf16(bstg[g][j].x, bstg[g][j].y);
                pk.y = pack_bf16(bstg[g][j].z, bstg[g][j].w);
                *(uint2*)&lsB[buf][g][ldsOffB0 + j * 32 * LDSTR] = pk;
            }
    };

    auto compute = [&](int buf) {
        const unsigned short* As = lsA[buf];
        Frag a[2];
        #pragma unroll
        for (int mA = 0; mA < 2; ++mA) {
            a[mA].u[0] = *(const uint4*)&As[fragA[mA]];
            a[mA].u[1] = *(const uint4*)&As[fragA[mA] + 8];
        }
        // Register-double-buffered B fragments: issue loads for pair p+1
        // before the WMMAs of pair p so waits don't drain all of LDS.
        Frag bb[2];
        {
            const unsigned short* Bs = lsB[buf][0];
            bb[0].u[0] = *(const uint4*)&Bs[fragB[0]];
            bb[0].u[1] = *(const uint4*)&Bs[fragB[0] + 8];
        }
        #pragma unroll
        for (int p = 0; p < 8; ++p) {
            const int g = p >> 1, nB = p & 1;
            if (p + 1 < 8) {
                const int gn = (p + 1) >> 1, nn = (p + 1) & 1;
                const unsigned short* Bs = lsB[buf][gn];
                bb[(p + 1) & 1].u[0] = *(const uint4*)&Bs[fragB[nn]];
                bb[(p + 1) & 1].u[1] = *(const uint4*)&Bs[fragB[nn] + 8];
            }
            acc[g][0][nB] = __builtin_amdgcn_wmma_f32_16x16x32_bf16(
                false, a[0].v, false, bb[p & 1].v, (short)0, acc[g][0][nB],
                false, false);
            acc[g][1][nB] = __builtin_amdgcn_wmma_f32_16x16x32_bf16(
                false, a[1].v, false, bb[p & 1].v, (short)0, acc[g][1][nB],
                false, false);
        }
    };

    load_stage(0);
    store_stage(0);
    __syncthreads();
    for (int kt = 0; kt < KITERS; ++kt) {
        int cur = kt & 1;
        if (kt + 1 < KITERS) load_stage(kt + 1);   // overlap global loads with WMMA
        compute(cur);
        if (kt + 1 < KITERS) store_stage(cur ^ 1);
        __syncthreads();
    }

    // ---- fused LSTM epilogue --------------------------------------------
    const unsigned HWb = (unsigned)(BATCH * DH) * 4u;   // byte offset of c_new
    #pragma unroll
    for (int mA = 0; mA < 2; ++mA) {
        #pragma unroll
        for (int nB = 0; nB < 2; ++nB) {
            unsigned N = (unsigned)(n0 + wn * 32 + nB * 16 + l);
            #pragma unroll
            for (int r = 0; r < 8; ++r) {
                unsigned M = (unsigned)(m0 + wm * 32 + mA * 16 + r + 8 * hi);
                unsigned off = (M * DH + N) * 4u;       // bytes
                float xi = acc[0][mA][nB][r] + bsum[0][nB];
                float xf = acc[1][mA][nB][r] + bsum[1][nB];
                float xg = acc[2][mA][nB][r] + bsum[2][nB];
                float xo = acc[3][mA][nB][r] + bsum[3][nB];
                float it = fsigmoid(xi);
                float ft = fsigmoid(xf);
                float gt = ftanh(xg);
                float ot = fsigmoid(xo);
                float cv = *(const float*)((const char*)c + (size_t)off);
                float cn = ft * cv + it * gt;
                float hn = ot * ftanh(cn);
                *(float*)((char*)out + (size_t)off)         = hn;  // h_new
                *(float*)((char*)out + (size_t)(HWb + off)) = cn;  // c_new
            }
        }
    }
}

extern "C" void kernel_launch(void* const* d_in, const int* in_sizes, int n_in,
                              void* d_out, int out_size, void* d_ws, size_t ws_size,
                              hipStream_t stream) {
    (void)in_sizes; (void)n_in; (void)out_size; (void)d_ws; (void)ws_size;
    const float* x = (const float*)d_in[0];
    const float* h = (const float*)d_in[1];
    const float* c = (const float*)d_in[2];
    // dict order: W_i2i,b_i2i, W_h2i,b_h2i, W_i2f,b_i2f, W_h2f,b_h2f,
    //             W_i2g,b_i2g, W_h2g,b_h2g, W_i2o,b_i2o, W_h2o,b_h2o
    const float* Wi[4] = {(const float*)d_in[3],  (const float*)d_in[7],
                          (const float*)d_in[11], (const float*)d_in[15]};
    const float* bi[4] = {(const float*)d_in[4],  (const float*)d_in[8],
                          (const float*)d_in[12], (const float*)d_in[16]};
    const float* Wh[4] = {(const float*)d_in[5],  (const float*)d_in[9],
                          (const float*)d_in[13], (const float*)d_in[17]};
    const float* bh[4] = {(const float*)d_in[6],  (const float*)d_in[10],
                          (const float*)d_in[14], (const float*)d_in[18]};
    float* out = (float*)d_out;

    dim3 grid(DH / TILE_N, BATCH / TILE_M);   // (16, 32)
    dim3 block(256);
    lstm_cell_wmma<<<grid, block, 0, stream>>>(
        x, h, c,
        Wi[0], Wi[1], Wi[2], Wi[3],
        Wh[0], Wh[1], Wh[2], Wh[3],
        bi[0], bi[1], bi[2], bi[3],
        bh[0], bh[1], bh[2], bh[3],
        out);
}
